// GBSR_SLightGCN_52458730553677
// MI455X (gfx1250) — matrix operand order repaired and checked by reading, not compiled, not run on hardware
//
#include <hip/hip_runtime.h>
#include <math.h>

typedef __attribute__((ext_vector_type(16))) _Float16 v16h;
typedef __attribute__((ext_vector_type(8)))  float    v8f;
typedef __attribute__((ext_vector_type(4)))  float    v4f;

#define NU 100000
#define NI 50000
#define NN 150000
#define DD 64
#define EE 1250000
#define SS 400000
#define ND 9600000   // NN * DD

// ---------------------------------------------------------------------------
// Init: write ego = [user; item] into BOTH output sum regions (layer-0 term).
// ---------------------------------------------------------------------------
__global__ void k_init_out(const float* __restrict__ user,
                           const float* __restrict__ item,
                           float* __restrict__ out) {
    long i = (long)blockIdx.x * blockDim.x + threadIdx.x;
    if (i >= (long)ND) return;
    float v = (i < (long)NU * DD) ? user[i] : item[i - (long)NU * DD];
    out[i] = v;
    out[(long)ND + i] = v;
}

__global__ void k_zero(float* __restrict__ p) {
    long i = ((long)blockIdx.x * blockDim.x + threadIdx.x) * 4;
    if (i + 3 < (long)ND) *(v4f*)(p + i) = (v4f){0.f, 0.f, 0.f, 0.f};
}

__global__ void k_add(float* __restrict__ sum, const float* __restrict__ buf) {
    long i = ((long)blockIdx.x * blockDim.x + threadIdx.x) * 4;
    if (i + 3 < (long)ND) {
        v4f s = *(v4f*)(sum + i);
        v4f b = *(const v4f*)(buf + i);
        s += b;
        *(v4f*)(sum + i) = s;
    }
}

// ---------------------------------------------------------------------------
// SpMM: out[rows[e]] += vals[e] * x[cols[e]].  One wave32 per edge:
// lane handles dims {2*lane, 2*lane+1} -> 256B coalesced gather + 64 atomics.
// Layer 1 reads the virtual ego = [user; item] without materializing it.
// ---------------------------------------------------------------------------
__global__ void __launch_bounds__(256) k_spmm(
    const int* __restrict__ rows, const int* __restrict__ cols,
    const float* __restrict__ vals, const float* __restrict__ x,
    const float* __restrict__ user, const float* __restrict__ item,
    int xIsEgo, float* __restrict__ out) {
    long t = (long)blockIdx.x * blockDim.x + threadIdx.x;
    long e = t >> 5;
    int lane = (int)(t & 31);
    if (e >= (long)EE) return;
    int r = rows[e], c = cols[e];
    float v = vals[e];
    const float* xr;
    if (xIsEgo)
        xr = (c < NU) ? (user + (long)c * DD) : (item + (long)(c - NU) * DD);
    else
        xr = x + (long)c * DD;
    float2 d = *(const float2*)(xr + lane * 2);
    float* o = out + (long)r * DD + lane * 2;
    atomicAdd(o,     v * d.x);
    atomicAdd(o + 1, v * d.y);
}

// ---------------------------------------------------------------------------
// Gate MLP over 16-edge tiles using v_wmma_f32_16x16x32_f16.
//   A (16x128) = cat(ego[su], ego[sv]) staged to LDS as f16
//   B (128x64) = W1, preloaded into 16 VGPR fragments once per wave
//   16 WMMAs per tile -> relu -> H(16x64) @ W2 -> gumbel-sigmoid gate
// Writes masked_vals[si] = sw * gate; accumulates sum(gate) for the mean.
// ---------------------------------------------------------------------------
__global__ void __launch_bounds__(256) k_gate(
    const float* __restrict__ user, const float* __restrict__ item,
    const int* __restrict__ rows, const int* __restrict__ cols,
    const float* __restrict__ vals, const int* __restrict__ sidx,
    const float* __restrict__ eps, const float* __restrict__ W1,
    const float* __restrict__ b1, const float* __restrict__ W2,
    const float* __restrict__ b2, float* __restrict__ mvals,
    float* __restrict__ gsum) {
    __shared__ _Float16 sW1h[128 * 64];            // 16 KB
    __shared__ float sW2[64];
    __shared__ float sb1[64];
    __shared__ float sb2;
    __shared__ int sNode[8][32];                   // per-wave node ids
    __shared__ unsigned char sTile[8][4096];       // per-wave: A(f16) then H(f32)

    const int tid  = threadIdx.x;
    const int lane = tid & 31;
    const int wave = tid >> 5;
    for (int i = tid; i < 128 * 64; i += 256) sW1h[i] = (_Float16)W1[i];
    for (int i = tid; i < 64; i += 256) { sW2[i] = W2[i]; sb1[i] = b1[i]; }
    if (tid == 0) sb2 = b2[0];
    __syncthreads();

    const int grp = lane >> 4;   // 0: lanes 0-15, 1: lanes 16-31
    const int l16 = lane & 15;

    // Preload all 16 B fragments of W1 (K-chunk x N-tile), kept in VGPRs.
    // B 32x16 layout: lane -> N=l16; K = (v<4 ? 2v : 16+2(v-4)) + grp*8 + p per half element.
    v16h bf[16];
#pragma unroll
    for (int kb = 0; kb < 4; ++kb)
#pragma unroll
        for (int nt = 0; nt < 4; ++nt)
#pragma unroll
            for (int v = 0; v < 8; ++v)
#pragma unroll
                for (int p = 0; p < 2; ++p) {
                    int kl = ((v < 4) ? (2 * v) : (16 + 2 * (v - 4))) + grp * 8 + p;
                    bf[kb * 4 + nt][2 * v + p] =
                        sW1h[(kb * 32 + kl) * 64 + nt * 16 + l16];
                }

    _Float16* catH = (_Float16*)sTile[wave];
    float*    Hb   = (float*)sTile[wave];

    const int nTiles = SS / 16;                    // 25000
    const int gw = blockIdx.x * 8 + wave;
    const int nw = gridDim.x * 8;
    for (int tile = gw; tile < nTiles; tile += nw) {
        const int base = tile * 16;
        int   si   = sidx[base + l16];
        int   node = grp ? cols[si] : rows[si];
        float sw   = vals[si];
        sNode[wave][lane] = node;
        asm volatile("s_wait_dscnt 0x0" ::: "memory");   // wave-local LDS vis

        // Stage 16x128 cat_emb tile as f16 (coalesced along k).
        for (int q = lane; q < 16 * 128; q += 32) {
            int j  = q >> 7;
            int k  = q & 127;
            int nd = sNode[wave][((k >> 6) << 4) + j];
            int kk = k & 63;
            float x = (nd < NU) ? user[(long)nd * DD + kk]
                                : item[(long)(nd - NU) * DD + kk];
            catH[j * 128 + k] = (_Float16)x;
        }
        asm volatile("s_wait_dscnt 0x0" ::: "memory");

        // C init with bias: C layout: N = l16 (+n0), M = grp*8 + vgpr-row.
        v8f acc[4];
#pragma unroll
        for (int nt = 0; nt < 4; ++nt) {
            float bv = sb1[nt * 16 + l16];
#pragma unroll
            for (int r = 0; r < 8; ++r) acc[nt][r] = bv;
        }
        // A 16x32 layout: M=l16; K = (v<4 ? 2v : 16+2(v-4)) + grp*8 + p.
#pragma unroll
        for (int kb = 0; kb < 4; ++kb) {
            v16h a;
#pragma unroll
            for (int v = 0; v < 8; ++v) {
                int kbv = (v < 4) ? (2 * v) : (16 + 2 * (v - 4));
#pragma unroll
                for (int p = 0; p < 2; ++p)
                    a[2 * v + p] = catH[l16 * 128 + kb * 32 + kbv + grp * 8 + p];
            }
#pragma unroll
            for (int nt = 0; nt < 4; ++nt)
                acc[nt] = __builtin_amdgcn_wmma_f32_16x16x32_f16(
                    false, a, false, bf[kb * 4 + nt], (short)0, acc[nt],
                    false, false);
        }
        // ReLU and write H (16x64 f32) over the tile buffer.
#pragma unroll
        for (int nt = 0; nt < 4; ++nt) {
            int n = nt * 16 + l16;
#pragma unroll
            for (int r = 0; r < 8; ++r) {
                float h = acc[nt][r];
                Hb[(grp * 8 + r) * 64 + n] = h > 0.f ? h : 0.f;
            }
        }
        asm volatile("s_wait_dscnt 0x0" ::: "memory");

        float g = 0.f;
        if (lane < 16) {
            float lg = sb2;
#pragma unroll 8
            for (int n = 0; n < 64; ++n) lg += Hb[lane * 64 + n] * sW2[n];
            float ep  = eps[base + lane] * (1.f - 2e-6f) + 1e-6f;
            float gum = logf(ep) - log1pf(-ep);
            float z   = (lg + gum) * 5.f;               // 1/TAU
            g = 1.f / (1.f + expf(-z)) + 0.5f;
            mvals[si] = sw * g;
        }
#pragma unroll
        for (int m = 16; m >= 1; m >>= 1) g += __shfl_xor(g, m, 32);
        if (lane == 0) atomicAdd(gsum, g);
    }
}

__global__ void k_final(float* __restrict__ out, const float* __restrict__ gs) {
    long i = (long)blockIdx.x * blockDim.x + threadIdx.x;
    if (i < 2L * ND) out[i] *= 0.25f;
    if (i == 0) out[2L * ND] = gs[0] / (float)SS;
}

// ---------------------------------------------------------------------------
extern "C" void kernel_launch(void* const* d_in, const int* in_sizes, int n_in,
                              void* d_out, int out_size, void* d_ws, size_t ws_size,
                              hipStream_t stream) {
    const float* user = (const float*)d_in[0];
    const float* item = (const float*)d_in[1];
    const int*   rows = (const int*)d_in[2];
    const int*   cols = (const int*)d_in[3];
    const float* vals = (const float*)d_in[4];
    const int*   sidx = (const int*)d_in[5];
    const float* eps  = (const float*)d_in[6];
    const float* W1   = (const float*)d_in[7];
    const float* b1   = (const float*)d_in[8];
    const float* W2   = (const float*)d_in[9];
    const float* b2   = (const float*)d_in[10];
    float* out = (float*)d_out;

    float* ws = (float*)d_ws;
    float* P1 = ws;
    float* P2 = ws + (size_t)ND;
    float* M1 = ws + 2 * (size_t)ND;
    float* M2 = ws + 3 * (size_t)ND;
    float* mvals = ws + 4 * (size_t)ND;
    float* gsum  = ws + 4 * (size_t)ND + (size_t)EE;

    hipMemsetAsync(gsum, 0, sizeof(float), stream);
    hipMemcpyAsync(mvals, vals, (size_t)EE * sizeof(float),
                   hipMemcpyDeviceToDevice, stream);

    k_gate<<<256, 256, 0, stream>>>(user, item, rows, cols, vals, sidx, eps,
                                    W1, b1, W2, b2, mvals, gsum);
    k_init_out<<<(ND + 255) / 256, 256, 0, stream>>>(user, item, out);

    float* sumP = out;
    float* sumM = out + (size_t)ND;

    const unsigned zb = (ND / 4 + 255) / 256;
    const unsigned sb = (unsigned)(((size_t)EE * 32 + 255) / 256);

    // layer 1 (x = virtual ego)
    k_zero<<<zb, 256, 0, stream>>>(P1);
    k_spmm<<<sb, 256, 0, stream>>>(rows, cols, vals,  user, user, item, 1, P1);
    k_add <<<zb, 256, 0, stream>>>(sumP, P1);
    k_zero<<<zb, 256, 0, stream>>>(M1);
    k_spmm<<<sb, 256, 0, stream>>>(rows, cols, mvals, user, user, item, 1, M1);
    k_add <<<zb, 256, 0, stream>>>(sumM, M1);
    // layer 2
    k_zero<<<zb, 256, 0, stream>>>(P2);
    k_spmm<<<sb, 256, 0, stream>>>(rows, cols, vals,  P1, user, item, 0, P2);
    k_add <<<zb, 256, 0, stream>>>(sumP, P2);
    k_zero<<<zb, 256, 0, stream>>>(M2);
    k_spmm<<<sb, 256, 0, stream>>>(rows, cols, mvals, M1, user, item, 0, M2);
    k_add <<<zb, 256, 0, stream>>>(sumM, M2);
    // layer 3 (reuse P1/M1)
    k_zero<<<zb, 256, 0, stream>>>(P1);
    k_spmm<<<sb, 256, 0, stream>>>(rows, cols, vals,  P2, user, item, 0, P1);
    k_add <<<zb, 256, 0, stream>>>(sumP, P1);
    k_zero<<<zb, 256, 0, stream>>>(M1);
    k_spmm<<<sb, 256, 0, stream>>>(rows, cols, mvals, M2, user, item, 0, M1);
    k_add <<<zb, 256, 0, stream>>>(sumM, M1);

    k_final<<<(2 * ND + 255) / 256, 256, 0, stream>>>(out, gsum);
}
